// RelativeAttention2d_50577534877995
// MI455X (gfx1250) — compile-verified, matrix operand order checked by reading, not run on hardware
//
#include <hip/hip_runtime.h>
#include <hip/hip_bf16.h>

// ---------------------------------------------------------------------------
// RelativeAttention2d for MI455X (gfx1250), f16 WMMA with f32 accumulation.
// B=16, C=D=256, H=32, N=1024, HEADS=8, HS=32.
// ---------------------------------------------------------------------------

typedef __attribute__((ext_vector_type(16))) _Float16 v16h;
typedef __attribute__((ext_vector_type(8)))  _Float16 v8h;
typedef __attribute__((ext_vector_type(8)))  float    v8f;

#define BATCH   16
#define NCH     256
#define NTOK    1024
#define NHEAD   8
#define HS      32
#define PER_B   (NCH * NTOK)          // 262144 elements per batch
#define REL_W   3969                  // (2*32-1)^2

static __device__ __forceinline__ v8f wmma_f16(v16h a, v16h b, v8f c) {
    return __builtin_amdgcn_wmma_f32_16x16x32_f16(false, a, false, b,
                                                  (short)0, c, false, false);
}

// ---------------------------------------------------------------------------
// Kernel 0: f32 -> f16 weight conversion
// ---------------------------------------------------------------------------
__global__ void cvt_f16_kernel(const float* __restrict__ src,
                               _Float16* __restrict__ dst, int n) {
    int i = blockIdx.x * 256 + threadIdx.x;
    if (i < n) dst[i] = (_Float16)src[i];
}

// ---------------------------------------------------------------------------
// Kernel 1: LayerNorm over channels. One wave per (b, spatial p).
// Output kept in the ORIGINAL [C][N] flat layout so the raw reshape to
// [N, C] rows is free.
// ---------------------------------------------------------------------------
__global__ void ln_kernel(const float* __restrict__ x,
                          const float* __restrict__ g,
                          const float* __restrict__ be,
                          _Float16* __restrict__ xnh) {
    int wid  = (blockIdx.x * 256 + threadIdx.x) >> 5;   // 0 .. B*N-1
    int lane = threadIdx.x & 31;
    int b = wid >> 10, p = wid & 1023;
    const float* xb = x + (size_t)b * PER_B + p;
    float v[8], s = 0.f, s2 = 0.f;
#pragma unroll
    for (int j = 0; j < 8; ++j) {
        v[j] = xb[(size_t)(lane + 32 * j) * NTOK];
        s += v[j]; s2 += v[j] * v[j];
    }
#pragma unroll
    for (int off = 16; off > 0; off >>= 1) {
        s  += __shfl_xor(s,  off);
        s2 += __shfl_xor(s2, off);
    }
    float mu = s * (1.f / 256.f);
    float var = s2 * (1.f / 256.f) - mu * mu;
    float rstd = rsqrtf(var + 1e-5f);
    _Float16* o = xnh + (size_t)b * PER_B + p;
#pragma unroll
    for (int j = 0; j < 8; ++j) {
        int c = lane + 32 * j;
        o[(size_t)c * NTOK] = (_Float16)((v[j] - mu) * rstd * g[c] + be[c]);
    }
}

// ---------------------------------------------------------------------------
// Kernel 2: bias table  tab[h][n][m] = rel_bias[h][rel_idx[n*N+m]]
// ---------------------------------------------------------------------------
__global__ void bias_tab_kernel(const float* __restrict__ rel_bias,
                                const int* __restrict__ rel_idx,
                                float* __restrict__ tab) {
    int i = blockIdx.x * 256 + threadIdx.x;     // 8M elements
    int h = i >> 20, nm = i & 1048575;
    tab[i] = rel_bias[h * REL_W + rel_idx[nm]];
}

// ---------------------------------------------------------------------------
// Kernel 3: QKV projections.  One wave = one 16x64 tile (A fragment reused
// across 4 B tiles).  q/k stored token-major per head ([h][m][32]) so the
// attention score WMMA gets contiguous per-lane fragments; v stored in
// natural flat layout (which IS the [h][s][1024] head view).
// ---------------------------------------------------------------------------
__global__ void qkv_kernel(const _Float16* __restrict__ xnh,
                           const _Float16* __restrict__ wqh,
                           const _Float16* __restrict__ wkh,
                           const _Float16* __restrict__ wvh,
                           const float* __restrict__ bq,
                           const float* __restrict__ bk,
                           const float* __restrict__ bv,
                           _Float16* __restrict__ qT,
                           _Float16* __restrict__ kT,
                           _Float16* __restrict__ vbuf) {
    int gw   = (blockIdx.x * 256 + threadIdx.x) >> 5;
    int lane = threadIdx.x & 31;
    int lo = lane & 15, hi = lane >> 4;
    int tile = gw & 255;                        // 64 n-tiles x 4 d-blocks
    int rem  = gw >> 8;
    int b    = rem & 15;
    int proj = rem >> 4;                        // 0=q 1=k 2=v (wave-uniform)
    int d0 = (tile & 3) << 6;                   // 64-wide d block
    int n0 = (tile >> 2) << 4;
    const _Float16* w    = (proj == 0) ? wqh : (proj == 1) ? wkh : wvh;
    const float*    bias = (proj == 0) ? bq  : (proj == 1) ? bk  : bv;
    const _Float16* A = xnh + (size_t)b * PER_B;
    __builtin_prefetch(A + (size_t)(n0 + lo) * 256, 0, 1);
    __builtin_prefetch(w + (size_t)(d0 + lo) * 256, 0, 1);

    v8f acc[4] = {};
#pragma unroll
    for (int k0 = 0; k0 < 256; k0 += 32) {
        v16h a = *(const v16h*)(A + (size_t)(n0 + lo) * 256 + k0 + hi * 16);
#pragma unroll
        for (int t = 0; t < 4; ++t) {
            v16h bf = *(const v16h*)(w + (size_t)(d0 + t * 16 + lo) * 256
                                       + k0 + hi * 16);
            acc[t] = wmma_f16(a, bf, acc[t]);
        }
    }
#pragma unroll
    for (int t = 0; t < 4; ++t) {
        int d = d0 + t * 16 + lo;
        float bia = bias[d];
        if (proj == 2) {
            _Float16* dst = vbuf + (size_t)b * PER_B;
#pragma unroll
            for (int r = 0; r < 8; ++r) {
                int n = n0 + hi * 8 + r;
                dst[(size_t)n * 256 + d] = (_Float16)(acc[t][r] + bia);
            }
        } else {
            _Float16* dst = ((proj == 0) ? qT : kT) + (size_t)b * PER_B;
#pragma unroll
            for (int r = 0; r < 8; ++r) {
                int n = n0 + hi * 8 + r;
                int h = n >> 7, s = (n >> 2) & 31, m = ((n & 3) << 8) + d;
                dst[(size_t)h * 32768 + m * 32 + s] = (_Float16)(acc[t][r] + bia);
            }
        }
    }
}

// ---------------------------------------------------------------------------
// Kernel 4: attention. One workgroup (8 waves) = one (b, h, 32-query block).
//  Phase 1: S[32x1024] = (Qt·K)/16 + bias  -> LDS (Q fragment hoisted)
//  Phase 2: softmax over keys (per row) in LDS
//  Phase 3: outT[32x32] = P · Vt; each wave covers both d-tiles per A-load,
//           K split 4 ways across waves, combined through LDS, packed f16
//           16B stores.
// ---------------------------------------------------------------------------
#define P_STRIDE 1028
#define ATTN_LDS_FLOATS (32 * P_STRIDE + 16 * 256)

__global__ void attn_kernel(const _Float16* __restrict__ qT,
                            const _Float16* __restrict__ kT,
                            const _Float16* __restrict__ vbuf,
                            const float* __restrict__ tab,
                            _Float16* __restrict__ obuf) {
    extern __shared__ float lds[];
    float* P   = lds;                     // 32 x 1028 f32
    float* red = lds + 32 * P_STRIDE;     // 8 waves x 2 tiles x 256 f32

    int lane = threadIdx.x & 31, wave = threadIdx.x >> 5;
    int lo = lane & 15, hi = lane >> 4;
    int mb = blockIdx.x & 31;
    int h  = (blockIdx.x >> 5) & 7;
    int b  = blockIdx.x >> 8;
    int m_base = mb << 5;

    const _Float16* q  = qT   + (size_t)b * PER_B + (size_t)h * 32768;
    const _Float16* k  = kT   + (size_t)b * PER_B + (size_t)h * 32768;
    const _Float16* v  = vbuf + (size_t)b * PER_B + (size_t)h * 32768;
    const float*    tb = tab  + (size_t)h * 1048576;
    __builtin_prefetch(k, 0, 1);
    __builtin_prefetch(v, 0, 1);

    // ---- Phase 1: scores into LDS (Q fragment is loop-invariant) ------
    {
        int mt  = wave & 1;               // which 16-row half of the m block
        int nt0 = wave >> 1;              // 0..3, strided n-tile walk
        int m0  = m_base + mt * 16;
        v16h a = *(const v16h*)(q + (size_t)(m0 + lo) * 32 + hi * 16);
        float* prow0 = P + (size_t)(mt * 16 + hi * 8) * P_STRIDE;
#pragma unroll 1
        for (int i = 0; i < 16; ++i) {
            int nt = nt0 + i * 4;         // covers all 64 n-tiles over 8 waves
            int n0 = nt * 16;
            v16h bf = *(const v16h*)(k + (size_t)(n0 + lo) * 32 + hi * 16);
            v8f acc = {};
            acc = wmma_f16(a, bf, acc);
            int n = n0 + lo;              // this lane's key column
            const float* trow = tb + (size_t)n * 1024 + m0 + hi * 8;
            float4 b0 = *(const float4*)trow;
            float4 b1 = *(const float4*)(trow + 4);
            float bias8[8] = {b0.x, b0.y, b0.z, b0.w, b1.x, b1.y, b1.z, b1.w};
            float* prow = prow0 + n;
#pragma unroll
            for (int r = 0; r < 8; ++r)
                prow[(size_t)r * P_STRIDE] = acc[r] * 0.0625f + bias8[r];
        }
    }
    __syncthreads();

    // ---- Phase 2: softmax over keys (row-wise) ------------------------
#pragma unroll 1
    for (int rr = 0; rr < 4; ++rr) {
        float* pr = P + (size_t)(wave * 4 + rr) * P_STRIDE;
        float mx = -1e30f;
        for (int j = lane; j < 1024; j += 32) mx = fmaxf(mx, pr[j]);
#pragma unroll
        for (int off = 16; off > 0; off >>= 1)
            mx = fmaxf(mx, __shfl_xor(mx, off));
        float sum = 0.f;
        for (int j = lane; j < 1024; j += 32) {
            float e = __expf(pr[j] - mx);
            pr[j] = e; sum += e;
        }
#pragma unroll
        for (int off = 16; off > 0; off >>= 1) sum += __shfl_xor(sum, off);
        float inv = 1.f / sum;
        for (int j = lane; j < 1024; j += 32) pr[j] *= inv;
    }
    __syncthreads();

    // ---- Phase 3: outT = P · Vt (both d-tiles per A-load) -------------
    {
        int mt = wave & 1, kh = wave >> 1;      // kh = 0..3, K quarter
        int m0t = mt * 16;
        v8f acc0 = {}, acc1 = {};
#pragma unroll 1
        for (int k0 = kh * 256; k0 < kh * 256 + 256; k0 += 32) {
            const float* prow = P + (size_t)(m0t + lo) * P_STRIDE + k0 + hi * 16;
            v16h a;
#pragma unroll
            for (int j = 0; j < 16; ++j) a[j] = (_Float16)prow[j];
            v16h bf0 = *(const v16h*)(v + (size_t)lo * 1024 + k0 + hi * 16);
            v16h bf1 = *(const v16h*)(v + (size_t)(16 + lo) * 1024 + k0 + hi * 16);
            acc0 = wmma_f16(a, bf0, acc0);
            acc1 = wmma_f16(a, bf1, acc1);
        }
        float* my = red + (size_t)(wave * 2) * 256 + lane * 8;
#pragma unroll
        for (int r = 0; r < 8; ++r) { my[r] = acc0[r]; my[256 + r] = acc1[r]; }
    }
    __syncthreads();

    if (wave < 4) {                             // wave -> (mt, dt) output tile
        int mt = wave & 1, dt = wave >> 1;
        v8h o;
#pragma unroll
        for (int r = 0; r < 8; ++r) {
            float s = 0.f;
#pragma unroll
            for (int kh = 0; kh < 4; ++kh)
                s += red[(size_t)((kh * 2 + mt) * 2 + dt) * 256 + lane * 8 + r];
            o[r] = (_Float16)s;
        }
        int d2 = dt * 16 + lo;
        int m2 = m_base + mt * 16 + hi * 8;
        *(v8h*)(obuf + (size_t)b * PER_B + (size_t)h * 32768
                + (size_t)d2 * 1024 + m2) = o;  // 8 consecutive f16 = 16B
    }
}

// ---------------------------------------------------------------------------
// Kernel 5: output projection + residual. One wave = one 16x64 tile with A
// fragment reuse. Flat attn buffer IS [p][dd] row-major; result element
// (p,c) lands at flat p*256+c which IS [C,H,H].
// ---------------------------------------------------------------------------
__global__ void proj_kernel(const _Float16* __restrict__ attno,
                            const _Float16* __restrict__ woh,
                            const float* __restrict__ bo,
                            const float* __restrict__ x,
                            float* __restrict__ out) {
    int gw   = (blockIdx.x * 256 + threadIdx.x) >> 5;
    int lane = threadIdx.x & 31;
    int lo = lane & 15, hi = lane >> 4;
    int tile = gw & 255;
    int b    = gw >> 8;
    int c0 = (tile & 3) << 6;                   // 64-wide c block
    int p0 = (tile >> 2) << 4;
    const _Float16* A = attno + (size_t)b * PER_B;
    __builtin_prefetch(A + (size_t)(p0 + lo) * 256, 0, 1);

    v8f acc[4] = {};
#pragma unroll
    for (int k0 = 0; k0 < 256; k0 += 32) {
        v16h a = *(const v16h*)(A + (size_t)(p0 + lo) * 256 + k0 + hi * 16);
#pragma unroll
        for (int t = 0; t < 4; ++t) {
            v16h bf = *(const v16h*)(woh + (size_t)(c0 + t * 16 + lo) * 256
                                         + k0 + hi * 16);
            acc[t] = wmma_f16(a, bf, acc[t]);
        }
    }
    const float* xb = x   + (size_t)b * PER_B;
    float*       ob = out + (size_t)b * PER_B;
#pragma unroll
    for (int t = 0; t < 4; ++t) {
        int c = c0 + t * 16 + lo;
        float bia = bo[c];
#pragma unroll
        for (int r = 0; r < 8; ++r) {
            int p = p0 + hi * 8 + r;
            size_t f = (size_t)p * 256 + c;
            ob[f] = xb[f] + acc[t][r] + bia;
        }
    }
}

// ---------------------------------------------------------------------------
extern "C" void kernel_launch(void* const* d_in, const int* in_sizes, int n_in,
                              void* d_out, int out_size, void* d_ws, size_t ws_size,
                              hipStream_t stream) {
    const float* x        = (const float*)d_in[0];
    const float* ln_g     = (const float*)d_in[1];
    const float* ln_b     = (const float*)d_in[2];
    const float* wq       = (const float*)d_in[3];
    const float* bq       = (const float*)d_in[4];
    const float* wk       = (const float*)d_in[5];
    const float* bk       = (const float*)d_in[6];
    const float* wv       = (const float*)d_in[7];
    const float* bv       = (const float*)d_in[8];
    const float* wo       = (const float*)d_in[9];
    const float* bo       = (const float*)d_in[10];
    const float* rel_bias = (const float*)d_in[11];
    const int*   rel_idx  = (const int*)d_in[12];
    float* out = (float*)d_out;

    // ---- workspace carve-up (~76 MB) ----------------------------------
    _Float16* xnh = (_Float16*)d_ws;
    _Float16* qT  = xnh + (size_t)BATCH * PER_B;
    _Float16* kT  = qT  + (size_t)BATCH * PER_B;
    _Float16* vb  = kT  + (size_t)BATCH * PER_B;
    _Float16* ob  = vb  + (size_t)BATCH * PER_B;
    _Float16* wqh = ob  + (size_t)BATCH * PER_B;
    _Float16* wkh = wqh + 65536;
    _Float16* wvh = wkh + 65536;
    _Float16* woh = wvh + 65536;
    float*    tab = (float*)(woh + 65536);      // 8 * 1024 * 1024 f32

    cvt_f16_kernel<<<256, 256, 0, stream>>>(wq, wqh, 65536);
    cvt_f16_kernel<<<256, 256, 0, stream>>>(wk, wkh, 65536);
    cvt_f16_kernel<<<256, 256, 0, stream>>>(wv, wvh, 65536);
    cvt_f16_kernel<<<256, 256, 0, stream>>>(wo, woh, 65536);

    ln_kernel<<<2048, 256, 0, stream>>>(x, ln_g, ln_b, xnh);
    bias_tab_kernel<<<32768, 256, 0, stream>>>(rel_bias, rel_idx, tab);
    qkv_kernel<<<1536, 256, 0, stream>>>(xnh, wqh, wkh, wvh, bq, bk, bv,
                                         qT, kT, vb);
    attn_kernel<<<4096, 256, ATTN_LDS_FLOATS * sizeof(float), stream>>>(
        qT, kT, vb, tab, ob);
    proj_kernel<<<512, 256, 0, stream>>>(ob, woh, bo, x, out);
}
